// AtomTransformer_35150012351222
// MI455X (gfx1250) — matrix operand order repaired and checked by reading, not compiled, not run on hardware
//
#include <hip/hip_runtime.h>
#include <math.h>

#define NATOM  2048
#define CDIM   128
#define CPDIM  16
#define NH     8
#define HD     16
#define HIDDIM 256
#define WINR   64
#define NBLK   3

typedef __attribute__((ext_vector_type(16))) _Float16 v16h;
typedef __attribute__((ext_vector_type(8)))  float    v8f;

__device__ __forceinline__ v8f wmma_f16(v16h a, v16h b, v8f c) {
  return __builtin_amdgcn_wmma_f32_16x16x32_f16(false, a, false, b, (short)0, c, false, false);
}

__device__ __forceinline__ float sigmoidf_(float x) { return 1.0f / (1.0f + __expf(-x)); }

// A fragment: 16x32 f16 tile from LDS row-major (_Float16), leading dim ld.
// lanes 0-15: row = lane, K = 0..15
// lanes 16-31: row = lane-16, K = 16..31
__device__ __forceinline__ v16h frag_a(const _Float16* src, int ld, int lane) {
  const int m = lane & 15;
  const int klo = (lane & 16) ? 16 : 0;
  const _Float16* p = src + m * ld + klo;
  v16h a;
#pragma unroll
  for (int j = 0; j < 8; ++j) a[j] = p[j];
#pragma unroll
  for (int j = 0; j < 8; ++j) a[8 + j] = p[8 + j];
  return a;
}

// B fragment: 32x16 f16 tile from global f32 row-major weights, leading dim ld.
// lanes 0-15: col = lane, K = 0..15 ; lanes 16-31: col = lane-16, K = 16..31
__device__ __forceinline__ v16h frag_b(const float* __restrict__ w, int ld, int lane) {
  const int n = lane & 15;
  const int kb = (lane & 16) ? 16 : 0;
  const float* p = w + (size_t)kb * ld + n;
  v16h b;
#pragma unroll
  for (int j = 0; j < 16; ++j) b[j] = (_Float16)p[(size_t)j * ld];
  return b;
}

// ---------------------------------------------------------------------------
// AdaLN: out = sigmoid(LN(s)*ss @ gw + gb) * LN(a) + LN(s)*ss @ bw
// ---------------------------------------------------------------------------
__global__ __launch_bounds__(256) void adaln_kernel(
    const float* __restrict__ a, const float* __restrict__ s,
    const float* __restrict__ gw, const float* __restrict__ gb,
    const float* __restrict__ bw, const float* __restrict__ ss,
    float* __restrict__ out) {
  __shared__ float    anl[16][CDIM];
  __shared__ _Float16 snh[16][CDIM];
  const int row0 = blockIdx.x * 16;
  const int tid  = threadIdx.x;
  const int r  = tid >> 4;
  const int c0 = tid & 15;
  {
    const float* ap = a + (size_t)(row0 + r) * CDIM;
    const float* sp = s + (size_t)(row0 + r) * CDIM;
    float va[8], vs[8];
    float suma = 0.f, sums = 0.f;
#pragma unroll
    for (int j = 0; j < 8; ++j) {
      va[j] = ap[c0 + 16 * j]; suma += va[j];
      vs[j] = sp[c0 + 16 * j]; sums += vs[j];
    }
#pragma unroll
    for (int off = 1; off < 16; off <<= 1) {
      suma += __shfl_xor(suma, off, 16);
      sums += __shfl_xor(sums, off, 16);
    }
    const float mua = suma * (1.0f / CDIM), mus = sums * (1.0f / CDIM);
    float vva = 0.f, vvs = 0.f;
#pragma unroll
    for (int j = 0; j < 8; ++j) {
      float da = va[j] - mua; vva += da * da;
      float ds = vs[j] - mus; vvs += ds * ds;
    }
#pragma unroll
    for (int off = 1; off < 16; off <<= 1) {
      vva += __shfl_xor(vva, off, 16);
      vvs += __shfl_xor(vvs, off, 16);
    }
    const float ra = rsqrtf(vva * (1.0f / CDIM) + 1e-5f);
    const float rs = rsqrtf(vvs * (1.0f / CDIM) + 1e-5f);
#pragma unroll
    for (int j = 0; j < 8; ++j) {
      const int c = c0 + 16 * j;
      anl[r][c] = (va[j] - mua) * ra;
      snh[r][c] = (_Float16)(((vs[j] - mus) * rs) * ss[c]);
    }
  }
  __syncthreads();

  const int wave = tid >> 5, lane = tid & 31;
  const int n0 = wave * 16;
  v8f accG = {}, accB = {};
#pragma unroll
  for (int k0 = 0; k0 < CDIM; k0 += 32) {
    v16h af = frag_a(&snh[0][k0], CDIM, lane);
    accG = wmma_f16(af, frag_b(gw + (size_t)k0 * CDIM + n0, CDIM, lane), accG);
    accB = wmma_f16(af, frag_b(bw + (size_t)k0 * CDIM + n0, CDIM, lane), accB);
  }
  const int n  = n0 + (lane & 15);
  const int mh = (lane & 16) ? 8 : 0;
  const float gbn = gb[n];
#pragma unroll
  for (int rr = 0; rr < 8; ++rr) {
    const int m = mh + rr;
    const float gate = sigmoidf_(accG[rr] + gbn);
    out[(size_t)(row0 + m) * CDIM + n] = gate * anl[m][n] + accB[rr];
  }
}

// ---------------------------------------------------------------------------
// qkvg: [q | k | v | sigmoid(g)] = an @ [wq|wk|wv|wg]  (+bq on q)
// ---------------------------------------------------------------------------
__global__ __launch_bounds__(256) void qkvg_kernel(
    const float* __restrict__ an,
    const float* __restrict__ wq, const float* __restrict__ bq,
    const float* __restrict__ wk, const float* __restrict__ wv,
    const float* __restrict__ wgm, float* __restrict__ qkvg) {
  __shared__ _Float16 ah[16][CDIM];
  const int row0 = blockIdx.x * 16;
  const int tid  = threadIdx.x;
  for (int idx = tid; idx < 16 * CDIM; idx += 256)
    ah[idx >> 7][idx & 127] = (_Float16)an[(size_t)(row0 + (idx >> 7)) * CDIM + (idx & 127)];
  __syncthreads();
  const int wave = tid >> 5, lane = tid & 31;
  const int n0 = wave * 16;
  const int n  = n0 + (lane & 15);
  const int mh = (lane & 16) ? 8 : 0;
  const float* ws4[4] = {wq, wk, wv, wgm};
#pragma unroll
  for (int mtx = 0; mtx < 4; ++mtx) {
    const float* w = ws4[mtx];
    v8f acc = {};
#pragma unroll
    for (int k0 = 0; k0 < CDIM; k0 += 32)
      acc = wmma_f16(frag_a(&ah[0][k0], CDIM, lane),
                     frag_b(w + (size_t)k0 * CDIM + n0, CDIM, lane), acc);
    const float bias = (mtx == 0) ? bq[n] : 0.f;
#pragma unroll
    for (int rr = 0; rr < 8; ++rr) {
      const int m = mh + rr;
      float v = acc[rr] + bias;
      if (mtx == 3) v = sigmoidf_(v);
      qkvg[(size_t)(row0 + m) * (4 * CDIM) + mtx * CDIM + n] = v;
    }
  }
}

// ---------------------------------------------------------------------------
// Windowed attention with band-only pair bias; online softmax; writes g*o.
// One wave per head, 16 queries per workgroup, 9 key tiles of 16.
// ---------------------------------------------------------------------------
__global__ __launch_bounds__(256) void attn_kernel(
    const float* __restrict__ qkvg, const float* __restrict__ z,
    const float* __restrict__ zs, const float* __restrict__ zb,
    const float* __restrict__ wb, float* __restrict__ go) {
  __shared__ float    bias_s[16][16][NH];
  __shared__ _Float16 pbuf[NH][16][16];
  const int q0   = blockIdx.x * 16;
  const int tid  = threadIdx.x;
  const int wave = tid >> 5, lane = tid & 31;
  const int h = wave;
  const int qi_b = tid >> 4, ki_b = tid & 15;

  const int lm  = lane & 15;
  const int klo = (lane & 16) ? 8 : 0;
  const int mh  = (lane & 16) ? 8 : 0;

  // Q fragment: dims 0..15 real, 16..31 zero-padded
  v16h qf;
  {
    const float* qp = qkvg + (size_t)(q0 + lm) * (4 * CDIM) + h * HD;
#pragma unroll
    for (int j = 0; j < 16; ++j) qf[j] = (lane & 16) ? (_Float16)0.f : (_Float16)qp[j];
  }

  v8f oacc = {};
  float Mst[8], Lst[8];
#pragma unroll
  for (int rr = 0; rr < 8; ++rr) { Mst[rr] = -3.0e38f; Lst[rr] = 0.f; }
  const float inv_sqrt_d = 0.25f;

  for (int t = -4; t <= 4; ++t) {
    const int k0 = q0 + t * 16;
    if (k0 <= -16 || k0 >= NATOM) continue;

    // ---- band pair bias tile: LN(z)+affine, project to 8 heads (all threads)
    {
      const int qg = q0 + qi_b;
      const int kg = k0 + ki_b;
      const int kc = kg < 0 ? 0 : (kg >= NATOM ? NATOM - 1 : kg);
      const float* zp = z + ((size_t)qg * NATOM + kc) * CPDIM;
      float zv[CPDIM];
      float sum = 0.f;
#pragma unroll
      for (int c = 0; c < CPDIM; ++c) { zv[c] = zp[c]; sum += zv[c]; }
      const float mu = sum * (1.0f / CPDIM);
      float var = 0.f;
#pragma unroll
      for (int c = 0; c < CPDIM; ++c) { float d = zv[c] - mu; var += d * d; }
      const float rstd = rsqrtf(var * (1.0f / CPDIM) + 1e-5f);
      float zn[CPDIM];
#pragma unroll
      for (int c = 0; c < CPDIM; ++c) zn[c] = (zv[c] - mu) * rstd * zs[c] + zb[c];
#pragma unroll
      for (int hh = 0; hh < NH; ++hh) {
        float b = 0.f;
#pragma unroll
        for (int c = 0; c < CPDIM; ++c) b += zn[c] * wb[c * NH + hh];
        bias_s[qi_b][ki_b][hh] = b;
      }
    }
    __syncthreads();

    // ---- scores: S = Q * K^T  (single WMMA, K dim zero-padded)
    v16h kf;
    if (lane < 16) {
      const int kg = k0 + lm;
      const int kc = kg < 0 ? 0 : (kg >= NATOM ? NATOM - 1 : kg);
      const float* kp = qkvg + (size_t)kc * (4 * CDIM) + CDIM + h * HD;
#pragma unroll
      for (int j = 0; j < 16; ++j) kf[j] = (_Float16)kp[j];
    } else {
#pragma unroll
      for (int j = 0; j < 16; ++j) kf[j] = (_Float16)0.f;
    }
    v8f sacc = {};
    sacc = wmma_f16(qf, kf, sacc);

    // ---- online softmax update
    const int kg_l = k0 + lm;
#pragma unroll
    for (int rr = 0; rr < 8; ++rr) {
      const int m  = mh + rr;
      const int qg = q0 + m;
      float sval = sacc[rr] * inv_sqrt_d + bias_s[m][lm][h];
      const int dd = kg_l - qg;
      const bool ok = (kg_l >= 0) && (kg_l < NATOM) && (dd >= -WINR) && (dd <= WINR);
      sval = ok ? sval : -1.0e30f;
      float rmax = sval;
#pragma unroll
      for (int off = 1; off < 16; off <<= 1) rmax = fmaxf(rmax, __shfl_xor(rmax, off, 16));
      const float newM  = fmaxf(Mst[rr], rmax);
      const float scale = __expf(Mst[rr] - newM);
      const float p     = __expf(sval - newM);
      float rsum = p;
#pragma unroll
      for (int off = 1; off < 16; off <<= 1) rsum += __shfl_xor(rsum, off, 16);
      Lst[rr] = Lst[rr] * scale + rsum;
      oacc[rr] *= scale;
      Mst[rr] = newM;
      pbuf[h][m][lm] = (_Float16)p;
    }

    // ---- O += P * V (single WMMA, K dim zero-padded)
    v16h pf;
#pragma unroll
    for (int j = 0; j < 16; ++j) pf[j] = (lane & 16) ? (_Float16)0.f : pbuf[h][lm][j];

    v16h vf;
    if (lane < 16) {
#pragma unroll
      for (int j = 0; j < 16; ++j) {
        const int kg = k0 + j;
        const int kc = kg < 0 ? 0 : (kg >= NATOM ? NATOM - 1 : kg);
        vf[j] = (_Float16)qkvg[(size_t)kc * (4 * CDIM) + 2 * CDIM + h * HD + lm];
      }
    } else {
#pragma unroll
      for (int j = 0; j < 16; ++j) vf[j] = (_Float16)0.f;
    }
    oacc = wmma_f16(pf, vf, oacc);
    __syncthreads();
  }

  // ---- epilogue: (o / L) * g  -> go
#pragma unroll
  for (int rr = 0; rr < 8; ++rr) {
    const int m = mh + rr;
    const float g = qkvg[(size_t)(q0 + m) * (4 * CDIM) + 3 * CDIM + h * HD + lm];
    go[(size_t)(q0 + m) * CDIM + h * HD + lm] = g * (oacc[rr] / Lst[rr]);
  }
}

// ---------------------------------------------------------------------------
// SwiGLU: hid = silu(tn @ w1) * (tn @ w2)
// ---------------------------------------------------------------------------
__global__ __launch_bounds__(256) void swiglu_kernel(
    const float* __restrict__ tn, const float* __restrict__ w1,
    const float* __restrict__ w2, float* __restrict__ hid) {
  __shared__ _Float16 th[16][CDIM];
  const int row0 = blockIdx.x * 16;
  const int tid  = threadIdx.x;
  for (int idx = tid; idx < 16 * CDIM; idx += 256)
    th[idx >> 7][idx & 127] = (_Float16)tn[(size_t)(row0 + (idx >> 7)) * CDIM + (idx & 127)];
  __syncthreads();
  const int wave = tid >> 5, lane = tid & 31;
  const int mh = (lane & 16) ? 8 : 0;
#pragma unroll
  for (int part = 0; part < 2; ++part) {
    const int n0 = (wave + part * 8) * 16;
    v8f acc1 = {}, acc2 = {};
#pragma unroll
    for (int k0 = 0; k0 < CDIM; k0 += 32) {
      v16h af = frag_a(&th[0][k0], CDIM, lane);
      acc1 = wmma_f16(af, frag_b(w1 + (size_t)k0 * HIDDIM + n0, HIDDIM, lane), acc1);
      acc2 = wmma_f16(af, frag_b(w2 + (size_t)k0 * HIDDIM + n0, HIDDIM, lane), acc2);
    }
    const int n = n0 + (lane & 15);
#pragma unroll
    for (int rr = 0; rr < 8; ++rr) {
      const int m = mh + rr;
      const float x = acc1[rr];
      hid[(size_t)(row0 + m) * HIDDIM + n] = (x * sigmoidf_(x)) * acc2[rr];
    }
  }
}

// ---------------------------------------------------------------------------
// Final: a = sig(s@wog+bog)*(go@wo) + sig(s@wsg+bsg)*(hid@w3)
// ---------------------------------------------------------------------------
__global__ __launch_bounds__(256) void final_kernel(
    const float* __restrict__ s, const float* __restrict__ go, const float* __restrict__ hid,
    const float* __restrict__ wo, const float* __restrict__ wog, const float* __restrict__ bog,
    const float* __restrict__ w3, const float* __restrict__ wsg, const float* __restrict__ bsg,
    float* __restrict__ aout) {
  __shared__ _Float16 sh[16][CDIM];
  __shared__ _Float16 goh[16][CDIM];
  __shared__ _Float16 hh[16][HIDDIM];
  const int row0 = blockIdx.x * 16;
  const int tid  = threadIdx.x;
  for (int idx = tid; idx < 16 * CDIM; idx += 256) {
    const int rr = idx >> 7, cc = idx & 127;
    sh[rr][cc]  = (_Float16)s[(size_t)(row0 + rr) * CDIM + cc];
    goh[rr][cc] = (_Float16)go[(size_t)(row0 + rr) * CDIM + cc];
  }
  for (int idx = tid; idx < 16 * HIDDIM; idx += 256) {
    const int rr = idx >> 8, cc = idx & 255;
    hh[rr][cc] = (_Float16)hid[(size_t)(row0 + rr) * HIDDIM + cc];
  }
  __syncthreads();
  const int wave = tid >> 5, lane = tid & 31;
  const int n0 = wave * 16;
  v8f accOG = {}, accO = {}, accSG = {}, accT = {};
#pragma unroll
  for (int k0 = 0; k0 < CDIM; k0 += 32) {
    v16h afs = frag_a(&sh[0][k0], CDIM, lane);
    v16h afg = frag_a(&goh[0][k0], CDIM, lane);
    accOG = wmma_f16(afs, frag_b(wog + (size_t)k0 * CDIM + n0, CDIM, lane), accOG);
    accSG = wmma_f16(afs, frag_b(wsg + (size_t)k0 * CDIM + n0, CDIM, lane), accSG);
    accO  = wmma_f16(afg, frag_b(wo  + (size_t)k0 * CDIM + n0, CDIM, lane), accO);
  }
#pragma unroll
  for (int k0 = 0; k0 < HIDDIM; k0 += 32) {
    v16h afh = frag_a(&hh[0][k0], HIDDIM, lane);
    accT = wmma_f16(afh, frag_b(w3 + (size_t)k0 * CDIM + n0, CDIM, lane), accT);
  }
  const int n  = n0 + (lane & 15);
  const int mh = (lane & 16) ? 8 : 0;
  const float bogn = bog[n], bsgn = bsg[n];
#pragma unroll
  for (int rr = 0; rr < 8; ++rr) {
    const int m = mh + rr;
    const float av = sigmoidf_(accOG[rr] + bogn) * accO[rr];
    const float tv = sigmoidf_(accSG[rr] + bsgn) * accT[rr];
    aout[(size_t)(row0 + m) * CDIM + n] = av + tv;
  }
}

// ---------------------------------------------------------------------------
extern "C" void kernel_launch(void* const* d_in, const int* in_sizes, int n_in,
                              void* d_out, int out_size, void* d_ws, size_t ws_size,
                              hipStream_t stream) {
  const float* A0  = (const float*)d_in[0];
  const float* S   = (const float*)d_in[1];
  const float* Z   = (const float*)d_in[2];
  const float* a_gw = (const float*)d_in[3];
  const float* a_gb = (const float*)d_in[4];
  const float* a_bw = (const float*)d_in[5];
  const float* a_ss = (const float*)d_in[6];
  const float* wq  = (const float*)d_in[7];
  const float* bq  = (const float*)d_in[8];
  const float* wk  = (const float*)d_in[9];
  const float* wv  = (const float*)d_in[10];
  const float* z_s = (const float*)d_in[11];
  const float* z_b = (const float*)d_in[12];
  const float* wb  = (const float*)d_in[13];
  const float* wg  = (const float*)d_in[14];
  const float* wo  = (const float*)d_in[15];
  const float* wog = (const float*)d_in[16];
  const float* bog = (const float*)d_in[17];
  const float* t_gw = (const float*)d_in[18];
  const float* t_gb = (const float*)d_in[19];
  const float* t_bw = (const float*)d_in[20];
  const float* t_ss = (const float*)d_in[21];
  const float* w1  = (const float*)d_in[22];
  const float* w2  = (const float*)d_in[23];
  const float* w3  = (const float*)d_in[24];
  const float* wsg = (const float*)d_in[25];
  const float* bsg = (const float*)d_in[26];

  const size_t NC = (size_t)NATOM * CDIM;
  float* ws   = (float*)d_ws;
  float* an   = ws;
  float* tn   = ws + NC;
  float* qkvg = ws + 2 * NC;   // [N, 4C]
  float* go   = ws + 6 * NC;   // [N, C]
  float* hid  = ws + 7 * NC;   // [N, 2C]
  float* abuf = ws + 9 * NC;   // [N, C] ping buffer

  dim3 grid(NATOM / 16), blk(256);
  const size_t CC = (size_t)CDIM * CDIM;
  for (int i = 0; i < NBLK; ++i) {
    const float* a_in  = (i == 0) ? A0 : abuf;
    float*       a_out = (i == NBLK - 1) ? (float*)d_out : abuf;
    adaln_kernel<<<grid, blk, 0, stream>>>(a_in, S, a_gw + i * CC, a_gb + i * CDIM,
                                           a_bw + i * CC, a_ss + i * CDIM, an);
    adaln_kernel<<<grid, blk, 0, stream>>>(a_in, S, t_gw + i * CC, t_gb + i * CDIM,
                                           t_bw + i * CC, t_ss + i * CDIM, tn);
    qkvg_kernel<<<grid, blk, 0, stream>>>(an, wq + i * CC, bq + i * CDIM,
                                          wk + i * CC, wv + i * CC, wg + i * CC, qkvg);
    attn_kernel<<<grid, blk, 0, stream>>>(qkvg, Z, z_s + i * CPDIM, z_b + i * CPDIM,
                                          wb + (size_t)i * CPDIM * NH, go);
    swiglu_kernel<<<grid, blk, 0, stream>>>(tn, w1 + (size_t)i * CDIM * HIDDIM,
                                            w2 + (size_t)i * CDIM * HIDDIM, hid);
    final_kernel<<<grid, blk, 0, stream>>>(S, go, hid, wo + i * CC, wog + i * CC,
                                           bog + i * CDIM, w3 + (size_t)i * HIDDIM * CDIM,
                                           wsg + i * CC, bsg + i * CDIM, a_out);
  }
}
